// Conv_layer_14276471292174
// MI455X (gfx1250) — matrix operand order repaired and checked by reading, not compiled, not run on hardware
//
#include <hip/hip_runtime.h>
#include <hip/hip_bf16.h>

// ---------------------------------------------------------------------------
// Graph-conv as batched GEMM on CDNA5 WMMA (bf16 in, f32 accum).
//   BS=8, V=4096, N=16, C_IN=C_OUT=64, K = 17*64 = 1088
// K reordered to f' = s*64 + c so each gathered neighbor row is contiguous;
// weights permuted once to match. A-fragments gathered per-lane straight from
// the L2-resident bf16 feature map via addrspace(1) loads (global_load_b128,
// no FLAT/DScnt). Neighbor indices are pipelined TWO segments ahead so the
// index load latency is covered by a full segment of WMMAs; rows for the next
// segment are prefetched at WGP scope (inline-asm global_prefetch_b8).
// B fragments are double-buffered one K-step ahead.
// ---------------------------------------------------------------------------

typedef __attribute__((ext_vector_type(16))) __bf16 v16bf;
typedef __attribute__((ext_vector_type(8)))  __bf16 v8bf;
typedef __attribute__((ext_vector_type(8)))  float  v8f;

// explicit global-address-space views (force global_load/store, never FLAT)
typedef const v8bf      __attribute__((address_space(1)))* gv8bf;
typedef const long long __attribute__((address_space(1)))* gi64;
typedef float           __attribute__((address_space(1)))* gf32;

#define BSZ   8
#define NV    4096
#define NNBR  16
#define NSEG  17            // N + 1 (self + neighbors)
#define CIN   64
#define COUT  64
#define KTOT  (NSEG * CIN)  // 1088

#define CAT16(lo, hi) __builtin_shufflevector((lo), (hi), 0,1,2,3,4,5,6,7,8,9,10,11,12,13,14,15)

// WGP-scope prefetch: default SCOPE field (0) pulls into all cache levels.
__device__ __forceinline__ void prefetch_wgp(const void* p) {
  asm volatile("global_prefetch_b8 %0, off" :: "v"((unsigned long long)(uintptr_t)p));
}

// ---- prep 1: feature_map f32 -> bf16 --------------------------------------
__global__ void gc_cvt_feat(const float* __restrict__ src,
                            __hip_bfloat16* __restrict__ dst, int n) {
  int i = blockIdx.x * blockDim.x + threadIdx.x;
  if (i < n) dst[i] = __float2bfloat16(src[i]);
}

// ---- prep 2: weights f32 [o][c*17+s] -> bf16 Wt[o][s*64+c] ----------------
__global__ void gc_perm_w(const float* __restrict__ w,
                          __hip_bfloat16* __restrict__ wt) {
  int i = blockIdx.x * blockDim.x + threadIdx.x;   // i = o*KTOT + f'
  if (i >= COUT * KTOT) return;
  int o = i / KTOT, f = i % KTOT;
  int s = f >> 6, c = f & 63;                      // f' = s*64 + c
  wt[i] = __float2bfloat16(w[o * KTOT + c * NSEG + s]);
}

// ---- main GEMM kernel ------------------------------------------------------
// Block = 128 threads (4 waves, wave32). Each wave: 32 vertices x 64 C_OUT
// (2 M-tiles x 4 N-tiles = 8 v8f accumulators). No LDS.
__global__ __launch_bounds__(128)
void gc_gemm(const long long* __restrict__ nbr,      // (BS,V,16) int64
             const __bf16*    __restrict__ featB,    // (BS,V,64) bf16
             const __bf16*    __restrict__ wtB,      // (COUT,KTOT) bf16, f'-order
             const float*     __restrict__ bias,     // (COUT,)
             float*           __restrict__ out) {    // (BS,V,COUT) f32
  const int lane  = threadIdx.x & 31;
  const int wave  = threadIdx.x >> 5;
  const int b     = blockIdx.x >> 5;                 // 32 blocks per batch
  const int vtile = (blockIdx.x & 31) * 128 + wave * 32;

  const int n15  = lane & 15;
  const int hsel = lane >> 4;                 // lane-half K sub-chunk select
  const int aoffA = hsel ? 8  : 0;            // ISA 16-bit A 16x32 layout
  const int aoffB = hsel ? 24 : 16;

  v8f acc[2][4];
  v8f z = {};
#pragma unroll
  for (int mt = 0; mt < 2; ++mt)
#pragma unroll
    for (int t = 0; t < 4; ++t) acc[mt][t] = z;

  const size_t vb   = (size_t)b * NV + vtile; // first vertex of this wave
  const size_t bimg = (size_t)b * NV;         // batch image base (vertices)

  // ---- pipelined gather state ----
  // roff: rows of current segment s (ready)
  // nidx: raw neighbor indices for segment s+1 (load in flight / done)
  size_t    roff[2];
  long long nidx[2];
#pragma unroll
  for (int mt = 0; mt < 2; ++mt) {
    roff[mt] = (vb + mt * 16 + n15) * CIN;                       // self rows
    nidx[mt] = *(gi64)(uintptr_t)&nbr[(vb + mt * 16 + n15) * NNBR + 0];
  }

  // B fragment loader for K-step index ks (kg = (ks>>1)*64 + (ks&1)*32)
  const __bf16* bbase = wtB + (size_t)n15 * KTOT + hsel * 8;
  auto loadB = [&](v16bf* bf, int ks) {
    const int kg = (ks >> 1) * CIN + (ks & 1) * 32;
#pragma unroll
    for (int t = 0; t < 4; ++t) {
      const __bf16* bp = bbase + (size_t)t * 16 * KTOT + kg;
      v8bf blo = *(gv8bf)(uintptr_t)bp;
      v8bf bhi = *(gv8bf)(uintptr_t)(bp + 16);
      bf[t] = CAT16(blo, bhi);
    }
  };
  auto loadA = [&](v16bf* af, int k0) {
#pragma unroll
    for (int mt = 0; mt < 2; ++mt) {
      v8bf lo = *(gv8bf)(uintptr_t)(featB + roff[mt] + k0 + aoffA);
      v8bf hi = *(gv8bf)(uintptr_t)(featB + roff[mt] + k0 + aoffB);
      af[mt] = CAT16(lo, hi);
    }
  };

  v16bf bcur[4], bnext[4], afr[2];
  loadB(bcur, 0);                              // prologue: B for (s=0,h=0)

  for (int s = 0; s < NSEG; ++s) {
    // 1) issue raw index load for segment s+2 (slot clamped; tail loads are
    //    valid-but-unused, keeping the body branchless)
    long long nidx2[2];
    const int slot2 = (s + 1 < NNBR) ? (s + 1) : (NNBR - 1);
#pragma unroll
    for (int mt = 0; mt < 2; ++mt)
      nidx2[mt] = *(gi64)(uintptr_t)&nbr[(vb + mt * 16 + n15) * NNBR + slot2];

    // 2) consume indices loaded a full segment ago -> next rows + prefetch
    size_t nroff[2];
#pragma unroll
    for (int mt = 0; mt < 2; ++mt) {
      nroff[mt] = (bimg + (size_t)(int)nidx[mt]) * CIN;
      prefetch_wgp(featB + nroff[mt]);         // WGP-scope global_prefetch_b8
    }

    // ---- K-step h=0: compute with bcur, load bnext for h=1 ----
    loadB(bnext, s * 2 + 1);
    loadA(afr, 0);
#pragma unroll
    for (int t = 0; t < 4; ++t)
#pragma unroll
      for (int mt = 0; mt < 2; ++mt)
        acc[mt][t] = __builtin_amdgcn_wmma_f32_16x16x32_bf16(
            false, afr[mt], false, bcur[t], (short)0, acc[mt][t], false, false);
#pragma unroll
    for (int t = 0; t < 4; ++t) bcur[t] = bnext[t];

    // ---- K-step h=1: compute with bcur, load bnext for next segment ----
    loadB(bnext, (s + 1 < NSEG) ? (s + 1) * 2 : 0);    // clamp on last seg
    loadA(afr, 32);
#pragma unroll
    for (int t = 0; t < 4; ++t)
#pragma unroll
      for (int mt = 0; mt < 2; ++mt)
        acc[mt][t] = __builtin_amdgcn_wmma_f32_16x16x32_bf16(
            false, afr[mt], false, bcur[t], (short)0, acc[mt][t], false, false);
#pragma unroll
    for (int t = 0; t < 4; ++t) bcur[t] = bnext[t];

    // 3) advance pipeline
#pragma unroll
    for (int mt = 0; mt < 2; ++mt) { roff[mt] = nroff[mt]; nidx[mt] = nidx2[mt]; }
  }

  // epilogue: C/D layout -> row = vgpr + hsel*8, col = n15; bias + ReLU
  const int mh = hsel * 8;
#pragma unroll
  for (int t = 0; t < 4; ++t) {
    const float bv = bias[t * 16 + n15];
#pragma unroll
    for (int mt = 0; mt < 2; ++mt) {
#pragma unroll
      for (int r = 0; r < 8; ++r) {
        float x = acc[mt][t][r] + bv;
        x = fmaxf(x, 0.0f);
        *(gf32)(uintptr_t)&out[(vb + mt * 16 + mh + r) * COUT + t * 16 + n15] = x;
      }
    }
  }
}

// ---------------------------------------------------------------------------
extern "C" void kernel_launch(void* const* d_in, const int* in_sizes, int n_in,
                              void* d_out, int out_size, void* d_ws, size_t ws_size,
                              hipStream_t stream) {
  const long long* nbr   = (const long long*)d_in[0];   // int64 (BS,V,16)
  // d_in[1] = vertices (unused by the reference math)
  const float* feat      = (const float*)d_in[2];       // (BS,V,64)
  const float* weights   = (const float*)d_in[3];       // (64,1088)
  const float* bias      = (const float*)d_in[4];       // (64,)
  float* out             = (float*)d_out;

  __hip_bfloat16* featB = (__hip_bfloat16*)d_ws;                         // 4 MB
  __hip_bfloat16* wtB   = (__hip_bfloat16*)((char*)d_ws + (4u << 20));   // 139 KB

  // prep: convert features + permute/convert weights (stream-ordered)
  const int nf = BSZ * NV * CIN;                 // 2,097,152
  gc_cvt_feat<<<(nf + 255) / 256, 256, 0, stream>>>(feat, featB, nf);
  const int nw = COUT * KTOT;                    // 69,632
  gc_perm_w<<<(nw + 255) / 256, 256, 0, stream>>>(weights, wtB);

  // main GEMM: 256 blocks x 128 threads, no LDS
  dim3 grid(BSZ * NV / 128);                     // 256
  dim3 block(128);
  gc_gemm<<<grid, block, 0, stream>>>(nbr,
                                      (const __bf16*)featB,
                                      (const __bf16*)wtB,
                                      bias, out);
}
